// Model_81269371175393
// MI455X (gfx1250) — matrix operand order repaired and checked by reading, not compiled, not run on hardware
//
#include <hip/hip_runtime.h>
#include <math.h>

// ---- model dims ----
static constexpr int cB  = 128;
static constexpr int cT  = 256;
static constexpr int cDV = 1024;
static constexpr int cDA = 128;
static constexpr int cH  = 128;
static constexpr int cDF = 512;
static constexpr int cKC = 7;
static constexpr int cM  = cB * cT;       // 32768 rows

typedef __attribute__((ext_vector_type(16))) _Float16 v16h;
typedef __attribute__((ext_vector_type(8)))  _Float16 v8h;
typedef __attribute__((ext_vector_type(4)))  _Float16 v4h;
typedef __attribute__((ext_vector_type(8)))  float    v8f;
typedef __attribute__((ext_vector_type(4)))  float    v4f;

// ---------------------------------------------------------------------------
// WMMA helpers (CDNA5: D = A(16x32 f16) x B(32x16 f16) + C(16x16 f32))
// ---------------------------------------------------------------------------
__device__ __forceinline__ v8f wmma32f16(v16h a, v16h b, v8f c) {
  return __builtin_amdgcn_wmma_f32_16x16x32_f16(
      false, a, false, b, (short)0, c, false, false);
}

// A fragment 16x32 f16: lane%16 = row; lanes 0-15 hold
// K={k0..k0+15}; lanes 16-31 hold K={k0+16..k0+31} (matches contiguous B).
__device__ __forceinline__ v16h a_frag_f16(const _Float16* rowp, int k0, int lane) {
  const int off = k0 + ((lane & 16) ? 16 : 0);
  v8h lo = *(const v8h*)(rowp + off);
  v8h hi = *(const v8h*)(rowp + off + 8);
  v16h r;
#pragma unroll
  for (int i = 0; i < 8; ++i) { r[i] = lo[i]; r[i + 8] = hi[i]; }
  return r;
}

__device__ __forceinline__ v16h a_frag_f32(const float* rowp, int k0, int lane) {
  const int off = k0 + ((lane & 16) ? 16 : 0);
  v16h r;
#pragma unroll
  for (int i = 0; i < 8; ++i) {
    r[i]     = (_Float16)rowp[off + i];
    r[i + 8] = (_Float16)rowp[off + 8 + i];
  }
  return r;
}

// ---------------------------------------------------------------------------
// WMMA GEMM, double-buffered unroll-by-2 (K % 64 == 0, K >= 64): even/odd
// register sets, no rotation copies; A stream prefetched via global_prefetch.
// Bt is N x K f16 (transposed weight): B-fragment loads are contiguous 32B.
// Block = 256 threads (8 waves); wave -> 16(M) x 64(N) tile; grid (M/128, N/64).
// ---------------------------------------------------------------------------
template <bool A32, bool GELU, bool OUT16>
__global__ void gemm_wmma(const void* __restrict__ Ap, const _Float16* __restrict__ Bt,
                          const float* __restrict__ bias, void* __restrict__ Cp,
                          int K, int lda, int ldc) {
  const int lane = threadIdx.x & 31;
  const int w    = threadIdx.x >> 5;
  const int m0   = blockIdx.x * 128 + w * 16;
  const int n0   = blockIdx.y * 64;
  const int l15  = lane & 15;
  const int bsel = (lane & 16) ? 16 : 0;

  const float*    a32 = (const float*)Ap    + (size_t)(m0 + l15) * lda;
  const _Float16* a16 = (const _Float16*)Ap + (size_t)(m0 + l15) * lda;
  const _Float16* brows[4];
#pragma unroll
  for (int nt = 0; nt < 4; ++nt)
    brows[nt] = Bt + (size_t)(n0 + nt * 16 + l15) * K + bsel;

  auto loadA = [&](int k0) -> v16h {
    return A32 ? a_frag_f32(a32, k0, lane) : a_frag_f16(a16, k0, lane);
  };

  v8f acc[4] = {};
  v16h afA = loadA(0);
  v16h bfA[4];
#pragma unroll
  for (int nt = 0; nt < 4; ++nt) bfA[nt] = *(const v16h*)(brows[nt]);

  int k0 = 0;
  for (; k0 + 64 < K; k0 += 64) {
    v16h afB = loadA(k0 + 32);
    v16h bfB[4];
#pragma unroll
    for (int nt = 0; nt < 4; ++nt) bfB[nt] = *(const v16h*)(brows[nt] + k0 + 32);
    if (A32) __builtin_prefetch(a32 + k0 + 128, 0, 1);   // global_prefetch_b8
    else     __builtin_prefetch(a16 + k0 + 128, 0, 1);
#pragma unroll
    for (int nt = 0; nt < 4; ++nt) acc[nt] = wmma32f16(afA, bfA[nt], acc[nt]);
    afA = loadA(k0 + 64);
#pragma unroll
    for (int nt = 0; nt < 4; ++nt) bfA[nt] = *(const v16h*)(brows[nt] + k0 + 64);
#pragma unroll
    for (int nt = 0; nt < 4; ++nt) acc[nt] = wmma32f16(afB, bfB[nt], acc[nt]);
  }
  { // tail: chunks k0 and k0+32 (K - k0 == 64)
    v16h afB = loadA(k0 + 32);
    v16h bfB[4];
#pragma unroll
    for (int nt = 0; nt < 4; ++nt) bfB[nt] = *(const v16h*)(brows[nt] + k0 + 32);
#pragma unroll
    for (int nt = 0; nt < 4; ++nt) acc[nt] = wmma32f16(afA, bfA[nt], acc[nt]);
#pragma unroll
    for (int nt = 0; nt < 4; ++nt) acc[nt] = wmma32f16(afB, bfB[nt], acc[nt]);
  }

  const int hi = (lane & 16) ? 8 : 0;
#pragma unroll
  for (int nt = 0; nt < 4; ++nt) {
    const int   col = n0 + nt * 16 + l15;
    const float bv  = bias[col];
#pragma unroll
    for (int r = 0; r < 8; ++r) {
      float v = acc[nt][r] + bv;
      if (GELU) v = 0.5f * v * (1.0f + erff(v * 0.70710678118654752f));
      const size_t idx = (size_t)(m0 + r + hi) * ldc + col;
      if (OUT16) ((_Float16*)Cp)[idx] = (_Float16)v;
      else       ((float*)Cp)[idx] = v;
    }
  }
}

// ---------------------------------------------------------------------------
// Attention. Block = 128 threads (4 waves); each wave owns a 16-row q tile.
// Scores stream to a per-wave LDS slab; distance prior computed with v_exp_f32
// (TRANS co-executes with WMMA); softmax is LDS-resident; 1/rowsum folded
// into the ctx epilogue. kv holds [k|v] rows of 256 f16; vt = V^T (B,H,T).
// Dynamic LDS = 4 * (16*256*4 + 16*256*2 + 64) bytes.
// ---------------------------------------------------------------------------
__global__ void attn_kernel(const _Float16* __restrict__ q, const _Float16* __restrict__ kv,
                            const _Float16* __restrict__ vt, _Float16* __restrict__ ctx) {
  extern __shared__ char smem_raw[];
  const int tid  = threadIdx.x;
  const int lane = tid & 31;
  const int w    = tid >> 5;
  char* slab = smem_raw + (size_t)w * (16 * 256 * 4 + 16 * 256 * 2 + 64);
  float*    Sbuf = (float*)slab;                                // 16 x 256 f32
  _Float16* Pbuf = (_Float16*)(slab + 16 * 256 * 4);            // 16 x 256 f16
  float*    rsb  = (float*)(slab + 16 * 256 * 4 + 16 * 256 * 2);// 16 f32

  const int tile = blockIdx.x * 4 + w;
  const int b    = tile >> 4;
  const int q0   = (tile & 15) * 16;
  const int l15  = lane & 15;
  const int hi   = (lane & 16) ? 8 : 0;
  const int bsel = (lane & 16) ? 16 : 0;
  const float scale = 0.08838834764831845f;      // 1/sqrt(128)
  const float nl2e  = -0.5307378454481409f;      // -log2(e)/e

  // Q fragments, resident across all key tiles
  const _Float16* qrow = q + (size_t)(b * cT + q0 + l15) * cH;
  v16h qa[4];
#pragma unroll
  for (int c = 0; c < 4; ++c) qa[c] = a_frag_f16(qrow, c * 32, lane);

  // ---- scores -> LDS, double-buffered over key tiles ----
  const _Float16* kbase = kv + (size_t)(b * cT) * 256;          // k = cols [0,128)
  auto load_kf = [&](int j, v16h (&f)[4]) {
    const _Float16* krow = kbase + (size_t)(j * 16 + l15) * 256;
#pragma unroll
    for (int c = 0; c < 4; ++c) f[c] = *(const v16h*)(krow + c * 32 + bsel);
  };
  auto score_tile = [&](int j, v16h (&f)[4]) {
    v8f a0 = {}, a1 = {};                        // split accumulators: short RAW chains
    a0 = wmma32f16(qa[0], f[0], a0);
    a1 = wmma32f16(qa[1], f[1], a1);
    a0 = wmma32f16(qa[2], f[2], a0);
    a1 = wmma32f16(qa[3], f[3], a1);
    const int tk = j * 16 + l15;
#pragma unroll
    for (int r = 0; r < 8; ++r) {
      const int tq = q0 + r + hi;
      const int d  = (tq > tk) ? (tq - tk) : (tk - tq);
      const float adj = __builtin_amdgcn_exp2f((float)d * nl2e);  // v_exp_f32
      Sbuf[(r + hi) * 256 + tk] = (a0[r] + a1[r]) * scale + adj;
    }
  };
  v16h kfA[4], kfB[4];
  load_kf(0, kfA);
  for (int j = 0; j + 2 < 16; j += 2) {
    load_kf(j + 1, kfB);
    score_tile(j, kfA);
    load_kf(j + 2, kfA);
    score_tile(j + 1, kfB);
  }
  load_kf(15, kfB);
  score_tile(14, kfA);
  score_tile(15, kfB);
  asm volatile("s_wait_dscnt 0" ::: "memory");

  // ---- softmax: lane handles row l15, column half selected by lane bit 4 ----
  const int   half = (lane >> 4) & 1;
  const float* srow = Sbuf + l15 * 256 + half * 128;
  float m = -3.0e38f;
#pragma unroll 8
  for (int i = 0; i < 32; ++i) {
    v4f x = *(const v4f*)(srow + 4 * i);
    m = fmaxf(m, fmaxf(fmaxf(x[0], x[1]), fmaxf(x[2], x[3])));
  }
  m = fmaxf(m, __shfl_xor(m, 16, 32));
  float s = 0.0f;
  _Float16* prw = Pbuf + l15 * 256 + half * 128;
#pragma unroll 8
  for (int i = 0; i < 32; ++i) {
    v4f x = *(const v4f*)(srow + 4 * i);
    const float e0 = __expf(x[0] - m), e1 = __expf(x[1] - m);
    const float e2 = __expf(x[2] - m), e3 = __expf(x[3] - m);
    s += (e0 + e1) + (e2 + e3);
    v4h p; p[0] = (_Float16)e0; p[1] = (_Float16)e1; p[2] = (_Float16)e2; p[3] = (_Float16)e3;
    *(v4h*)(prw + 4 * i) = p;                 // unnormalized P
  }
  s += __shfl_xor(s, 16, 32);
  if (half == 0) rsb[l15] = 1.0f / s;
  asm volatile("s_wait_dscnt 0" ::: "memory");

  // ---- ctx = P @ V (K = 256 -> 8 chunks, N = 128 -> 8 tiles) ----
  v8f Cx[8] = {};
  const _Float16* pr = Pbuf + l15 * 256;
#pragma unroll
  for (int c = 0; c < 8; ++c) {
    v16h pa = a_frag_f16(pr, c * 32, lane);
    v16h bf[8];
#pragma unroll
    for (int nt = 0; nt < 8; ++nt)
      bf[nt] = *(const v16h*)(vt + (size_t)(b * cH + nt * 16 + l15) * cT + c * 32 + bsel);
#pragma unroll
    for (int nt = 0; nt < 8; ++nt) Cx[nt] = wmma32f16(pa, bf[nt], Cx[nt]);
  }
  float rr[8];
#pragma unroll
  for (int r = 0; r < 8; ++r) rr[r] = rsb[r + hi];
#pragma unroll
  for (int nt = 0; nt < 8; ++nt)
#pragma unroll
    for (int r = 0; r < 8; ++r)
      ctx[(size_t)(b * cT + q0 + r + hi) * cH + nt * 16 + l15] =
          (_Float16)(Cx[nt][r] * rr[r]);
}

// ---------------------------------------------------------------------------
// Helper kernels
// ---------------------------------------------------------------------------
// W (K x N f32, row-major) -> Wt (N x K f16)
__global__ void wconv_kernel(const float* __restrict__ W, _Float16* __restrict__ Wt,
                             int K, int N) {
  const int i = blockIdx.x * blockDim.x + threadIdx.x;
  if (i < K * N) {
    const int kk = i / N, n = i % N;
    Wt[(size_t)n * K + kk] = (_Float16)W[(size_t)kk * N + n];
  }
}

__global__ void biascat_kernel(const float* __restrict__ a, const float* __restrict__ b,
                               float* __restrict__ o) {
  const int i = threadIdx.x;
  o[i] = (i < cH) ? a[i] : b[i - cH];
}

// kv (B*T x 256 f16, v in cols 128..255) -> vt (B x H x T f16)
__global__ void vtrans_kernel(const _Float16* __restrict__ kv, _Float16* __restrict__ vt) {
  const int i = blockIdx.x * blockDim.x + threadIdx.x;
  if (i < cB * cT * cH) {
    const int b = i / (cT * cH);
    const int t = (i / cH) % cT;
    const int h = i % cH;
    vt[(size_t)(b * cH + h) * cT + t] = kv[(size_t)(b * cT + t) * 256 + cH + h];
  }
}

// h = LN(f_v + ca) * g + b ; one block (256 threads) per (b,t) row
__global__ void ln_kernel(const float* __restrict__ x, const _Float16* __restrict__ ca,
                          const float* __restrict__ g, const float* __restrict__ be,
                          _Float16* __restrict__ h16) {
  const int row = blockIdx.x;
  const int tid = threadIdx.x;
  const float*    xr  = x  + (size_t)row * (cDV + cDA);
  const _Float16* car = ca + (size_t)row * cDV;
  __shared__ float sh1[256], sh2[256];
  float vals[cDV / 256];
  float s = 0.0f, ss = 0.0f;
#pragma unroll
  for (int j = 0; j < cDV / 256; ++j) {
    const int i = tid + j * 256;
    const float v = xr[i] + (float)car[i];
    vals[j] = v; s += v; ss += v * v;
  }
  sh1[tid] = s; sh2[tid] = ss;
  __syncthreads();
  for (int o = 128; o > 0; o >>= 1) {
    if (tid < o) { sh1[tid] += sh1[tid + o]; sh2[tid] += sh2[tid + o]; }
    __syncthreads();
  }
  const float mu  = sh1[0] * (1.0f / cDV);
  const float var = sh2[0] * (1.0f / cDV) - mu * mu;
  const float rst = rsqrtf(var + 1e-5f);
#pragma unroll
  for (int j = 0; j < cDV / 256; ++j) {
    const int i = tid + j * 256;
    h16[(size_t)row * cDV + i] = (_Float16)((vals[j] - mu) * rst * g[i] + be[i]);
  }
}

// temporal conv (left pad 6) + sigmoid; block = one batch, h2 tile + Wc in LDS
__global__ void conv_sig_kernel(const float* __restrict__ h2, const float* __restrict__ Wc,
                                const float* __restrict__ bc, float* __restrict__ out) {
  extern __shared__ float csm[];
  float* sW = csm;                    // 896
  float* sh = csm + cH * cKC;         // 256*128
  const int b = blockIdx.x, tid = threadIdx.x;
  for (int i = tid; i < cH * cKC; i += 256) sW[i] = Wc[i];
  const float* hb = h2 + (size_t)b * cT * cH;
  for (int i = tid * 4; i < cT * cH; i += 256 * 4)
    *(v4f*)(sh + i) = *(const v4f*)(hb + i);
  __syncthreads();
  const int t = tid;
  float acc = bc[0];
#pragma unroll
  for (int j = 0; j < cKC; ++j) {
    const int tt = t + j - (cKC - 1);
    if (tt < 0) continue;
    const float* hr = sh + tt * cH;
    for (int hh = 0; hh < cH; ++hh) acc += hr[hh] * sW[hh * cKC + j];
  }
  out[b * cT + t] = 1.0f / (1.0f + __expf(-acc));
}

// ---------------------------------------------------------------------------
extern "C" void kernel_launch(void* const* d_in, const int* in_sizes, int n_in,
                              void* d_out, int out_size, void* d_ws, size_t ws_size,
                              hipStream_t stream) {
  (void)in_sizes; (void)n_in; (void)out_size; (void)ws_size;
  const float* x   = (const float*)d_in[0];
  const float* Wq  = (const float*)d_in[1];
  const float* bq  = (const float*)d_in[2];
  const float* Wk  = (const float*)d_in[3];
  const float* bk  = (const float*)d_in[4];
  const float* Wv  = (const float*)d_in[5];
  const float* bv  = (const float*)d_in[6];
  const float* Wo  = (const float*)d_in[7];
  const float* bo  = (const float*)d_in[8];
  const float* lng = (const float*)d_in[9];
  const float* lnb = (const float*)d_in[10];
  const float* W1  = (const float*)d_in[11];
  const float* b1  = (const float*)d_in[12];
  const float* W2  = (const float*)d_in[13];
  const float* b2  = (const float*)d_in[14];
  const float* Wc  = (const float*)d_in[15];
  const float* bc  = (const float*)d_in[16];
  float* out = (float*)d_out;

  char*  ws  = (char*)d_ws;
  size_t off = 0;
  auto alloc = [&](size_t bytes) -> void* {
    off = (off + 255) & ~((size_t)255);
    void* p = ws + off;
    off += bytes;
    return p;
  };
  _Float16* wqT  = (_Float16*)alloc((size_t)cH * cDA * 2);        // 128 x 128
  _Float16* wkvT = (_Float16*)alloc((size_t)2 * cH * cDV * 2);    // 256 x 1024
  _Float16* woT  = (_Float16*)alloc((size_t)cDV * cH * 2);        // 1024 x 128
  _Float16* w1T  = (_Float16*)alloc((size_t)cDF * cDV * 2);       // 512 x 1024
  _Float16* w2T  = (_Float16*)alloc((size_t)cH * cDF * 2);        // 128 x 512
  float*    kvb  = (float*)   alloc(256 * 4);
  _Float16* qb   = (_Float16*)alloc((size_t)cM * cH * 2);
  _Float16* kvbf = (_Float16*)alloc((size_t)cM * 256 * 2);        // [k|v]
  _Float16* vtb  = (_Float16*)alloc((size_t)cM * cH * 2);
  _Float16* ctx  = (_Float16*)alloc((size_t)cM * cH * 2);
  _Float16* ca   = (_Float16*)alloc((size_t)cM * cDV * 2);
  _Float16* h16  = (_Float16*)alloc((size_t)cM * cDV * 2);
  _Float16* g16  = (_Float16*)alloc((size_t)cM * cDF * 2);
  float*    h2   = (float*)   alloc((size_t)cM * cH * 4);

  const dim3 blk(256);

  // f16 transposed weights ([Wk|Wv] fused into one 256-col operand)
  wconv_kernel<<<(cDA * cH + 255) / 256, blk, 0, stream>>>(Wq, wqT, cDA, cH);
  wconv_kernel<<<(cDV * cH + 255) / 256, blk, 0, stream>>>(Wk, wkvT, cDV, cH);
  wconv_kernel<<<(cDV * cH + 255) / 256, blk, 0, stream>>>(Wv, wkvT + (size_t)cH * cDV, cDV, cH);
  wconv_kernel<<<(cH * cDV + 255) / 256, blk, 0, stream>>>(Wo, woT, cH, cDV);
  wconv_kernel<<<(cDV * cDF + 255) / 256, blk, 0, stream>>>(W1, w1T, cDV, cDF);
  wconv_kernel<<<(cDF * cH + 255) / 256, blk, 0, stream>>>(W2, w2T, cDF, cH);
  biascat_kernel<<<1, 256, 0, stream>>>(bk, bv, kvb);

  // q projection (A = f_a slice) ; fused k|v projection (reads f_v once)
  gemm_wmma<true,  false, true ><<<dim3(cM / 128, cH / 64),  blk, 0, stream>>>(
      (const void*)(x + cDV), wqT, bq, qb, cDA, cDV + cDA, cH);
  gemm_wmma<true,  false, true ><<<dim3(cM / 128, 256 / 64), blk, 0, stream>>>(
      (const void*)x, wkvT, kvb, kvbf, cDV, cDV + cDA, 256);

  vtrans_kernel<<<(cB * cT * cH + 255) / 256, blk, 0, stream>>>(kvbf, vtb);

  // attention: 4 waves/block, LDS = 4 slabs (scores f32 + P f16 + rowsums)
  const size_t attn_lds = 4 * (16 * 256 * 4 + 16 * 256 * 2 + 64);
  attn_kernel<<<cM / (16 * 4), dim3(128), attn_lds, stream>>>(qb, kvbf, vtb, ctx);

  // output projection
  gemm_wmma<false, false, true ><<<dim3(cM / 128, cDV / 64), blk, 0, stream>>>(
      (const void*)ctx, woT, bo, ca, cH, cH, cDV);

  // residual + layernorm
  ln_kernel<<<cM, blk, 0, stream>>>(x, ca, lng, lnb, h16);

  // MLP: gelu(h @ W1 + b1) @ W2 + b2
  gemm_wmma<false, true,  true ><<<dim3(cM / 128, cDF / 64), blk, 0, stream>>>(
      (const void*)h16, w1T, b1, g16, cDV, cDV, cDF);
  gemm_wmma<false, false, false><<<dim3(cM / 128, cH / 64),  blk, 0, stream>>>(
      (const void*)g16, w2T, b2, h2, cDF, cDF, cH);

  // temporal conv + sigmoid (h2 batch tile + weights staged in LDS)
  conv_sig_kernel<<<cB, blk, (cH * cKC + cT * cH) * 4, stream>>>(h2, Wc, bc, out);
}
